// MultiScaleDecoder_deterministic_84430467105495
// MI455X (gfx1250) — matrix operand order repaired and checked by reading, not compiled, not run on hardware
//
#include <hip/hip_runtime.h>
#include <hip/hip_bf16.h>

typedef __attribute__((ext_vector_type(16))) __bf16 v16bf;
typedef __attribute__((ext_vector_type(8)))  __bf16 v8bf;
typedef __attribute__((ext_vector_type(8)))  float  v8f;

#define Bsz  128
#define Dsz  64
#define Tsz  2048
#define Esz  128
#define HSsz 256
#define HFsz 512
#define KIN  320      // D + HS
#define KCAT 832      // D + HS + HF
#define G3F  1536     // 3*HF
#define G3S  768      // 3*HS

// ---- LDS layout (bytes) ----
// S_T / Hn_T are TRANSPOSED: element (row m, col n) lives at [n*16 + m].
#define OFF_S    0                         // S_T  [1536][16] f32 (raw gate sums)
#define OFF_HN   (OFF_S   + G3F*16*4)      // Hn_T [512][16]  f32 (raw hidden n-gate)
#define OFF_HF   (OFF_HN  + HFsz*16*4)     // h_fast [16][512] f32
#define OFF_HS   (OFF_HF  + 16*HFsz*4)     // h_slow [16][256] f32
#define OFF_ZP   (OFF_HS  + 16*HSsz*4)     // z_prev [16][64]  f32
#define OFF_GIS  (OFF_ZP  + 16*Dsz*4)      // gi_slow [16][768] f32 (time-invariant)
#define OFF_BIF  (OFF_GIS + 16*G3S*4)      // bih_f [1536] f32
#define OFF_BHF  (OFF_BIF + G3F*4)         // bhh_f [1536] f32
#define OFF_BHS  (OFF_BHF + G3F*4)         // bhh_s [768]  f32
#define OFF_BO   (OFF_BHS + G3S*4)         // b_out [64]   f32
#define OFF_ACT  (OFF_BO  + Dsz*4)         // act [16][832] bf16 = [z | h_slow | h_fast]
#define SMEM_BYTES (OFF_ACT + 16*KCAT*2)   // = 275712 B (< 320KB/WGP)

// ---- workspace layout (bytes) ----
#define WS_WF   0                                   // bf16 [1536][832]  = [Wih_f | Whh_f]
#define WS_WHS  (WS_WF  + (size_t)G3F*KCAT*2)       // bf16 [768][256]   = Whh_s
#define WS_WO   (WS_WHS + (size_t)G3S*HSsz*2)       // bf16 [64][512]    = W_out
#define WS_GIS  (WS_WO  + (size_t)Dsz*HFsz*2)       // f32  [128][768]   = e@Wih_s.T + bih_s
#define WS_TOTAL (WS_GIS + (size_t)Bsz*G3S*4)

// Fast gate nonlinearities: v_exp_f32 + v_rcp_f32 (no IEEE divide sequence)
__device__ __forceinline__ float sigm(float x) {
    return __builtin_amdgcn_rcpf(1.0f + __expf(-x));
}
__device__ __forceinline__ float tanh_fast(float x) {
    // tanh(x) = 2*sigmoid(2x) - 1
    return fmaf(2.0f, __builtin_amdgcn_rcpf(1.0f + __expf(-2.0f * x)), -1.0f);
}

// A-fragment: 16x32 bf16, contiguous K order (lane half 0: K 0-15; half 1: K 16-31)
__device__ __forceinline__ v16bf load_a(const __bf16* rowbase_k) {
    v8bf lo = *(const v8bf*)(rowbase_k);
    v8bf hi = *(const v8bf*)(rowbase_k + 8);
    return __builtin_shufflevector(lo, hi, 0,1,2,3,4,5,6,7,8,9,10,11,12,13,14,15);
}

// ---------- prep 1: fp32 weights -> bf16, concatenated fast weights ----------
__global__ void msd_prep_weights(const float* __restrict__ Wih_f, const float* __restrict__ Whh_f,
                                 const float* __restrict__ Whh_s, const float* __restrict__ W_out,
                                 __bf16* __restrict__ Wf, __bf16* __restrict__ Whs,
                                 __bf16* __restrict__ Wo) {
    const int NW1 = G3F * KCAT, NW2 = G3S * HSsz, NW3 = Dsz * HFsz;
    for (int i = blockIdx.x * blockDim.x + threadIdx.x; i < NW1 + NW2 + NW3;
         i += gridDim.x * blockDim.x) {
        if (i < NW1) {
            int n = i / KCAT, k = i - n * KCAT;
            float v = (k < KIN) ? Wih_f[n * KIN + k] : Whh_f[n * HFsz + (k - KIN)];
            Wf[i] = (__bf16)v;
        } else if (i < NW1 + NW2) {
            Whs[i - NW1] = (__bf16)Whh_s[i - NW1];
        } else {
            Wo[i - NW1 - NW2] = (__bf16)W_out[i - NW1 - NW2];
        }
    }
}

// ---------- prep 2: gi_slow = e @ Wih_s.T + bih_s (one-time, tiny) ----------
__global__ void msd_prep_gis(const float* __restrict__ e, const float* __restrict__ Wih_s,
                             const float* __restrict__ bih_s, float* __restrict__ gis) {
    int i = blockIdx.x * blockDim.x + threadIdx.x;
    if (i >= Bsz * G3S) return;
    int b = i / G3S, n = i - b * G3S;
    float acc = bih_s[n];
    for (int k = 0; k < Esz; ++k) acc += e[b * Esz + k] * Wih_s[n * Esz + k];
    gis[i] = acc;
}

// ---------- main persistent RNN kernel: 1 workgroup per 16 batch rows ----------
__global__ __launch_bounds__(512, 1) void msd_main(
    const float* __restrict__ zin, const int* __restrict__ use_gt, const int* __restrict__ Kp,
    const float* __restrict__ bhh_s, const float* __restrict__ bih_f,
    const float* __restrict__ bhh_f, const float* __restrict__ b_out,
    const __bf16* __restrict__ Wf, const __bf16* __restrict__ Whs,
    const __bf16* __restrict__ Wo, const float* __restrict__ gis,
    float* __restrict__ out) {
    extern __shared__ char smem[];
    float*  Sf   = (float*)(smem + OFF_S);    // transposed [n][m]
    float*  Hnf  = (float*)(smem + OFF_HN);   // transposed [n'][m]
    float*  hF   = (float*)(smem + OFF_HF);
    float*  hS   = (float*)(smem + OFF_HS);
    float*  zP   = (float*)(smem + OFF_ZP);
    float*  gisL = (float*)(smem + OFF_GIS);
    float*  biF  = (float*)(smem + OFF_BIF);
    float*  bhF  = (float*)(smem + OFF_BHF);
    float*  bhS  = (float*)(smem + OFF_BHS);
    float*  boL  = (float*)(smem + OFF_BO);
    __bf16* act  = (__bf16*)(smem + OFF_ACT);

    const int tid  = threadIdx.x;
    const int wave = tid >> 5;
    const int lane = tid & 31;
    const int l15  = lane & 15;
    const int half = lane >> 4;
    const int b0   = blockIdx.x * 16;
    const int Kv   = Kp[0];

    // ---- one-time staging: zero states, cache time-invariant operands in LDS ----
    for (int i = tid; i < 16 * HFsz; i += 512) hF[i] = 0.0f;
    for (int i = tid; i < 16 * HSsz; i += 512) hS[i] = 0.0f;
    for (int i = tid; i < 16 * Dsz;  i += 512) zP[i] = 0.0f;
    for (int i = tid; i < 16 * KCAT; i += 512) act[i] = (__bf16)0.0f;
    for (int i = tid; i < 16 * G3S;  i += 512) gisL[i] = gis[(size_t)b0 * G3S + i];
    for (int i = tid; i < G3F;       i += 512) { biF[i] = bih_f[i]; bhF[i] = bhh_f[i]; }
    for (int i = tid; i < G3S;       i += 512) bhS[i] = bhh_s[i];
    if (tid < Dsz) boL[tid] = b_out[tid];
    __syncthreads();

    int kc = 0;  // t % K without the modulo
    for (int t = 0; t < Tsz; ++t) {
        // ===== slow GRU update every K steps (used for steps t..t+K-1) =====
        if (kc == 0) {
#pragma clang loop unroll(disable)
            for (int i = 0; i < 3; ++i) {               // 16 waves * 3 tiles = 768 cols
                const int n0 = (wave * 3 + i) * 16;
                v8f acc = {};
                const __bf16* brow = Whs + (size_t)(n0 + l15) * HSsz + half * 16;
                for (int kb = 0; kb < HSsz / 32; ++kb) {
                    v16bf a  = load_a(act + l15 * KCAT + Dsz + kb * 32 + half * 16);
                    v16bf bm = *(const v16bf*)(brow + kb * 32);
                    acc = __builtin_amdgcn_wmma_f32_16x16x32_bf16(false, a, false, bm,
                                                                  (short)0, acc, false, false);
                }
                // raw gh -> S_T[n][m]; 8 consecutive floats per lane (2x ds_store_b128)
                *(v8f*)(Sf + (n0 + l15) * 16 + half * 8) = acc;
            }
            __syncthreads();
            for (int idx = tid; idx < 16 * HSsz; idx += 512) {
                const int m = idx >> 8, j = idx & (HSsz - 1);
                const float r  = sigm(gisL[m * G3S + j] + Sf[j * 16 + m] + bhS[j]);
                const float zg = sigm(gisL[m * G3S + HSsz + j] + Sf[(HSsz + j) * 16 + m]
                                      + bhS[HSsz + j]);
                const float hn = Sf[(2 * HSsz + j) * 16 + m] + bhS[2 * HSsz + j];
                const float nn = tanh_fast(gisL[m * G3S + 2 * HSsz + j] + r * hn);
                const float h  = (1.0f - zg) * nn + zg * hS[idx];
                hS[idx] = h;
                act[m * KCAT + Dsz + j] = (__bf16)h;               // bf16 copy for GEMMs
            }
            __syncthreads();
        }
        if (++kc == Kv) kc = 0;

        // ===== fast gates: gi (K=0..319 of act) and gh (K=320..831) fused =====
#pragma clang loop unroll(disable)
        for (int i = 0; i < 6; ++i) {                   // 16 waves * 6 tiles = 1536 cols
            const int n0 = (wave * 6 + i) * 16;
            v8f ai = {}, ah = {};
            const __bf16* arow = act + l15 * KCAT + half * 16;
            const __bf16* brow = Wf + (size_t)(n0 + l15) * KCAT + half * 16;
            for (int kb = 0; kb < KIN / 32; ++kb) {     // input-gate part
                v16bf a  = load_a(arow + kb * 32);
                v16bf bm = *(const v16bf*)(brow + kb * 32);
                ai = __builtin_amdgcn_wmma_f32_16x16x32_bf16(false, a, false, bm,
                                                             (short)0, ai, false, false);
            }
            for (int kb = KIN / 32; kb < KCAT / 32; ++kb) {  // hidden-gate part
                v16bf a  = load_a(arow + kb * 32);
                v16bf bm = *(const v16bf*)(brow + kb * 32);
                ah = __builtin_amdgcn_wmma_f32_16x16x32_bf16(false, a, false, bm,
                                                             (short)0, ah, false, false);
            }
            // combined raw gi+gh -> S_T; raw ah (n-third) -> Hn_T; biases deferred
            *(v8f*)(Sf + (n0 + l15) * 16 + half * 8) = ai + ah;
            if (n0 >= 2 * HFsz)
                *(v8f*)(Hnf + (n0 - 2 * HFsz + l15) * 16 + half * 8) = ah;
        }
        __syncthreads();

        // ===== fast elementwise GRU update =====
        for (int idx = tid; idx < 16 * HFsz; idx += 512) {
            const int m = idx >> 9, j = idx & (HFsz - 1);
            const float r  = sigm(Sf[j * 16 + m] + biF[j] + bhF[j]);
            const float zg = sigm(Sf[(HFsz + j) * 16 + m] + biF[HFsz + j] + bhF[HFsz + j]);
            const float ahn = Hnf[j * 16 + m];                  // raw hidden n-part
            const float hn  = ahn + bhF[2 * HFsz + j];          // h_n incl. bias
            const float in_ = Sf[(2 * HFsz + j) * 16 + m] - ahn + biF[2 * HFsz + j]; // i_n
            const float nn  = tanh_fast(in_ + r * hn);
            const float h   = (1.0f - zg) * nn + zg * hF[idx];
            hF[idx] = h;
            act[m * KCAT + KIN + j] = (__bf16)h;
        }
        __syncthreads();

        // ===== delta = h_fast @ W_out.T ; emit z_hat ; teacher-forced z update =====
        if (wave < 4) {                                  // 4 tiles cover 64 cols
            const int n0 = wave * 16;
            const int n  = n0 + l15;
            v8f acc = {};
            const __bf16* arow = act + l15 * KCAT + KIN + half * 16;
            const __bf16* brow = Wo + (size_t)(n0 + l15) * HFsz + half * 16;
            for (int kb = 0; kb < HFsz / 32; ++kb) {
                v16bf a  = load_a(arow + kb * 32);
                v16bf bm = *(const v16bf*)(brow + kb * 32);
                acc = __builtin_amdgcn_wmma_f32_16x16x32_bf16(false, a, false, bm,
                                                              (short)0, acc, false, false);
            }
            const float bo = boL[n];
            const int   gt = use_gt[t];
            for (int v = 0; v < 8; ++v) {
                const int m = v + half * 8;
                const float zh = zP[m * Dsz + n] + acc[v] + bo;
                const size_t g = ((size_t)(b0 + m) * Dsz + n) * Tsz + t;
                out[g] = zh;
                const float zn = (gt > 0) ? zin[g] : zh;
                zP[m * Dsz + n]  = zn;
                act[m * KCAT + n] = (__bf16)zn;
            }
        }
        __syncthreads();
    }
}

extern "C" void kernel_launch(void* const* d_in, const int* in_sizes, int n_in,
                              void* d_out, int out_size, void* d_ws, size_t ws_size,
                              hipStream_t stream) {
    const float* z     = (const float*)d_in[0];
    const float* e     = (const float*)d_in[1];
    const int*   ugt   = (const int*)d_in[2];
    const int*   Kp    = (const int*)d_in[3];
    const float* Wih_s = (const float*)d_in[4];
    const float* Whh_s = (const float*)d_in[5];
    const float* bih_s = (const float*)d_in[6];
    const float* bhh_s = (const float*)d_in[7];
    const float* Wih_f = (const float*)d_in[8];
    const float* Whh_f = (const float*)d_in[9];
    const float* bih_f = (const float*)d_in[10];
    const float* bhh_f = (const float*)d_in[11];
    const float* W_out = (const float*)d_in[12];
    const float* b_out = (const float*)d_in[13];
    float* out = (float*)d_out;

    char* ws = (char*)d_ws;
    __bf16* Wf  = (__bf16*)(ws + WS_WF);
    __bf16* Whs = (__bf16*)(ws + WS_WHS);
    __bf16* Wo  = (__bf16*)(ws + WS_WO);
    float*  gis = (float*)(ws + WS_GIS);

    (void)in_sizes; (void)n_in; (void)out_size; (void)ws_size;
    hipFuncSetAttribute((const void*)msd_main,
                        hipFuncAttributeMaxDynamicSharedMemorySize, SMEM_BYTES);

    {
        const int total = G3F * KCAT + G3S * HSsz + Dsz * HFsz;
        msd_prep_weights<<<(total + 255) / 256, 256, 0, stream>>>(Wih_f, Whh_f, Whh_s, W_out,
                                                                  Wf, Whs, Wo);
    }
    msd_prep_gis<<<(Bsz * G3S + 255) / 256, 256, 0, stream>>>(e, Wih_s, bih_s, gis);

    msd_main<<<dim3(Bsz / 16), dim3(512), SMEM_BYTES, stream>>>(
        z, ugt, Kp, bhh_s, bih_f, bhh_f, b_out, Wf, Whs, Wo, gis, out);
}
